// MultiHeadAttentionBlock_4698694222338
// MI455X (gfx1250) — matrix operand order repaired and checked by reading, not compiled, not run on hardware
//
#include <hip/hip_runtime.h>

#define DMODEL 1024
#define SEQ    2048
#define NB     2
#define NH     16
#define HD     64
#define NTOK   (NB * SEQ)          // 4096
#define SCALE  0.03125f            // 1/sqrt(1024)
#define NEGINF -1e30f
#define MASKNEG -1000000000.0f

typedef __bf16 bf16_t;
typedef __attribute__((ext_vector_type(16))) __bf16 v16bf;
typedef __attribute__((ext_vector_type(8)))  float  v8f;

// ---------- helpers ----------

__device__ __forceinline__ bf16_t f2bf(float f) {
  unsigned u = __float_as_uint(f);
  u += 0x7FFFu + ((u >> 16) & 1u);               // round-to-nearest-even
  union { unsigned short s; bf16_t b; } cv;
  cv.s = (unsigned short)(u >> 16);
  return cv.b;
}

// A-matrix fragment (16x32 bf16, M x K): lane holds row m = lane%16.
// lanes 0-15: K bytes [0,16)+[32,48); lanes 16-31: [16,32)+[48,64)
__device__ __forceinline__ v16bf load_frag_a(const bf16_t* row, unsigned lane) {
  const char* p = (const char*)row + ((lane & 16u) ? 16 : 0);
  union { uint4 u[2]; v16bf v; } r;
  r.u[0] = *(const uint4*)(p);
  r.u[1] = *(const uint4*)(p + 32);
  return r.v;
}

// B-matrix fragment (32x16 bf16, K x N): lane holds column n = lane%16,
// lanes 0-15: K=[0,8)+[16,24); lanes 16-31: K=[8,16)+[24,32).
__device__ __forceinline__ v16bf load_frag_b(const bf16_t* col, unsigned lane) {
  const char* p = (const char*)col + ((lane & 16u) ? 16 : 0);
  union { uint4 u[2]; v16bf v; } r;
  r.u[0] = *(const uint4*)(p);
  r.u[1] = *(const uint4*)(p + 32);
  return r.v;
}

__device__ __forceinline__ v8f wmma_bf16(v16bf a, v16bf b, v8f c) {
  return __builtin_amdgcn_wmma_f32_16x16x32_bf16(false, a, false, b,
                                                 (short)0, c, false, false);
}

// xor-shuffle via ds_bpermute (wave32); masks 1/2/4/8 stay inside a 16-lane
// half, matching the C-matrix row distribution.
__device__ __forceinline__ float bperm_xor(float v, int m) {
  int lane = (int)(threadIdx.x & 31u);
  return __int_as_float(
      __builtin_amdgcn_ds_bpermute(((lane ^ m) << 2), __float_as_int(v)));
}

// ---------- fp32 -> bf16 conversion ----------

__global__ void cvt_bf16(const float* __restrict__ x, bf16_t* __restrict__ y, int n) {
  int i = blockIdx.x * blockDim.x + threadIdx.x;
  int stride = gridDim.x * blockDim.x;
  for (; i < n; i += stride) y[i] = f2bf(x[i]);
}

// ---------- WMMA GEMM: Y = X @ W^T  (M=NTOK, N=K=DMODEL) ----------
// mode 0: bf16 head-split [B,H,S,HD]; mode 1: bf16 transposed [B,H,HD,S];
// mode 2: fp32 token-major [M, DMODEL].
// block = 128 (4 waves); each wave computes a 32x64 slab (2 M-tiles x 4
// N-tiles, B reused across M). Register double-buffering overlaps the next
// k-step's 6 fragment loads with the current 8 WMMAs.
__global__ void __launch_bounds__(128)
gemm_wmma(const bf16_t* __restrict__ X, const bf16_t* __restrict__ W,
          void* __restrict__ Y, int mode) {
  unsigned lane = threadIdx.x & 31u;
  unsigned wave = threadIdx.x >> 5;
  int m0 = ((int)blockIdx.x << 7) + ((int)wave << 5);  // 32 rows per wave
  int n0 = (int)blockIdx.y << 6;                       // 64 cols per block

  const bf16_t* x0 = X + (size_t)(m0 + (int)(lane & 15u)) * DMODEL;
  const bf16_t* x1 = x0 + (size_t)16 * DMODEL;
  const bf16_t* w0 = W + (size_t)(n0 + (int)(lane & 15u)) * DMODEL;

  v8f acc[2][4] = {};

  v16bf a0 = load_frag_a(x0, lane);
  v16bf a1 = load_frag_a(x1, lane);
  v16bf bb[4];
#pragma unroll
  for (int n = 0; n < 4; ++n)
    bb[n] = load_frag_b(w0 + (size_t)n * 16 * DMODEL, lane);

#pragma unroll 2
  for (int kk = 0; kk < DMODEL - 32; kk += 32) {
    // issue next k-step's loads first so they overlap the WMMAs below
    v16bf na0 = load_frag_a(x0 + kk + 32, lane);
    v16bf na1 = load_frag_a(x1 + kk + 32, lane);
    v16bf nb[4];
#pragma unroll
    for (int n = 0; n < 4; ++n)
      nb[n] = load_frag_b(w0 + (size_t)n * 16 * DMODEL + kk + 32, lane);

#pragma unroll
    for (int n = 0; n < 4; ++n) {
      acc[0][n] = wmma_bf16(a0, bb[n], acc[0][n]);
      acc[1][n] = wmma_bf16(a1, bb[n], acc[1][n]);
    }
    a0 = na0; a1 = na1;
#pragma unroll
    for (int n = 0; n < 4; ++n) bb[n] = nb[n];
  }
#pragma unroll
  for (int n = 0; n < 4; ++n) {                        // tail k-step
    acc[0][n] = wmma_bf16(a0, bb[n], acc[0][n]);
    acc[1][n] = wmma_bf16(a1, bb[n], acc[1][n]);
  }

  int col = (int)(lane & 15u);
#pragma unroll
  for (int mt = 0; mt < 2; ++mt) {
    int rowBase = m0 + mt * 16 + ((lane & 16u) ? 8 : 0);
#pragma unroll
    for (int nt = 0; nt < 4; ++nt) {
      int n = n0 + nt * 16 + col;
#pragma unroll
      for (int r = 0; r < 8; ++r) {
        int m = rowBase + r;
        float v = acc[mt][nt][r];
        if (mode == 2) {
          ((float*)Y)[(size_t)m * DMODEL + n] = v;
        } else {
          int b = m >> 11, s = m & (SEQ - 1);
          int h = n >> 6,  hd = n & (HD - 1);
          bf16_t bv = f2bf(v);
          if (mode == 0)
            ((bf16_t*)Y)[((size_t)(b * NH + h) * SEQ + s) * HD + hd] = bv;
          else
            ((bf16_t*)Y)[((size_t)(b * NH + h) * HD + hd) * SEQ + s] = bv;
        }
      }
    }
  }
}

// ---------- fused flash attention ----------
// grid (SEQ/128, NH, NB); block 256 = 8 waves, each wave owns 16 queries.
__global__ void __launch_bounds__(256)
attn_wmma(const bf16_t* __restrict__ Qh, const bf16_t* __restrict__ Kh,
          const bf16_t* __restrict__ VT, const int* __restrict__ mask,
          bf16_t* __restrict__ ctx) {
  __shared__ __align__(16) bf16_t ptile[8][16 * 64];  // per-wave P staging
  unsigned lane = threadIdx.x & 31u;
  unsigned wave = threadIdx.x >> 5;
  int b = blockIdx.z, h = blockIdx.y;
  int q0 = ((int)blockIdx.x * 8 + (int)wave) * 16;

  const bf16_t* Qbh  = Qh + (size_t)(b * NH + h) * SEQ * HD;
  const bf16_t* Kbh  = Kh + (size_t)(b * NH + h) * SEQ * HD;
  const bf16_t* VTbh = VT + (size_t)(b * NH + h) * HD * SEQ;
  const int*    mb   = mask + (size_t)b * SEQ * SEQ;
  bf16_t* pt = &ptile[wave][0];

  v16bf aq0 = load_frag_a(Qbh + (size_t)(q0 + (int)(lane & 15u)) * HD,      lane);
  v16bf aq1 = load_frag_a(Qbh + (size_t)(q0 + (int)(lane & 15u)) * HD + 32, lane);

  v8f acc[4] = {};
  float mrow[8], lrow[8];
#pragma unroll
  for (int r = 0; r < 8; ++r) { mrow[r] = NEGINF; lrow[r] = 0.f; }
  int rowBase = q0 + ((lane & 16u) ? 8 : 0);

  for (int kc = 0; kc < SEQ; kc += 64) {
    // ---- batch-load all K fragments for this 64-key chunk ----
    v16bf bk0[4], bk1[4];
#pragma unroll
    for (int j = 0; j < 4; ++j) {
      const bf16_t* kr = Kbh + (size_t)(kc + j * 16 + (int)(lane & 15u)) * HD;
      bk0[j] = load_frag_b(kr,      lane);
      bk1[j] = load_frag_b(kr + 32, lane);
    }
    // prefetch next chunk's K rows (global_prefetch_b8)
    if (kc + 64 < SEQ) {
      __builtin_prefetch(Kbh + (size_t)(kc + 64 + (int)lane) * HD, 0, 1);
      __builtin_prefetch(Kbh + (size_t)(kc + 96 + (int)lane) * HD, 0, 1);
    }

    // ---- scores S = (Q K^T)/32, masked ----
    v8f sc[4];
    float tmax[8];
#pragma unroll
    for (int r = 0; r < 8; ++r) tmax[r] = NEGINF;
#pragma unroll
    for (int j = 0; j < 4; ++j) {
      int key = kc + j * 16 + (int)(lane & 15u);
      v8f c = {};
      c = wmma_bf16(aq0, bk0[j], c);
      c = wmma_bf16(aq1, bk1[j], c);
#pragma unroll
      for (int r = 0; r < 8; ++r) {
        float s = c[r] * SCALE;
        s = mb[(size_t)(rowBase + r) * SEQ + key] ? s : MASKNEG;
        c[r] = s;
        tmax[r] = fmaxf(tmax[r], s);
      }
      sc[j] = c;
    }

    // ---- online softmax update ----
#pragma unroll
    for (int r = 0; r < 8; ++r) {
      float t = tmax[r];
      t = fmaxf(t, bperm_xor(t, 1));
      t = fmaxf(t, bperm_xor(t, 2));
      t = fmaxf(t, bperm_xor(t, 4));
      t = fmaxf(t, bperm_xor(t, 8));
      float mn = fmaxf(mrow[r], t);
      float alpha = __expf(mrow[r] - mn);
      mrow[r] = mn;
      lrow[r] *= alpha;
      acc[0][r] *= alpha; acc[1][r] *= alpha;
      acc[2][r] *= alpha; acc[3][r] *= alpha;
    }

    // ---- P = exp(S - m): stash bf16 in LDS for the A-fragment transpose ----
#pragma unroll
    for (int j = 0; j < 4; ++j) {
#pragma unroll
      for (int r = 0; r < 8; ++r) {
        float p = __expf(sc[j][r] - mrow[r]);
        sc[j][r] = p;
        pt[(size_t)(r + ((lane & 16u) ? 8 : 0)) * 64 + j * 16 + (lane & 15u)] =
            f2bf(p);
      }
    }
#pragma unroll
    for (int r = 0; r < 8; ++r) {
      float s = sc[0][r] + sc[1][r] + sc[2][r] + sc[3][r];
      s += bperm_xor(s, 1); s += bperm_xor(s, 2);
      s += bperm_xor(s, 4); s += bperm_xor(s, 8);
      lrow[r] += s;
    }

    asm volatile("s_wait_dscnt 0" ::: "memory");  // P writes visible to DS reads

    // ---- acc += P @ V: batch-load all V^T fragments, then 8 WMMAs ----
    v16bf pa0 = load_frag_a(pt + (size_t)(lane & 15u) * 64,      lane);
    v16bf pa1 = load_frag_a(pt + (size_t)(lane & 15u) * 64 + 32, lane);
    const bf16_t* vt0 = VTbh + (size_t)(lane & 15u) * SEQ + kc;
    v16bf bv0[4], bv1[4];
#pragma unroll
    for (int n = 0; n < 4; ++n) {
      bv0[n] = load_frag_b(vt0 + (size_t)n * 16 * SEQ,      lane);
      bv1[n] = load_frag_b(vt0 + (size_t)n * 16 * SEQ + 32, lane);
    }
#pragma unroll
    for (int n = 0; n < 4; ++n) {
      acc[n] = wmma_bf16(pa0, bv0[n], acc[n]);
      acc[n] = wmma_bf16(pa1, bv1[n], acc[n]);
    }
  }

  // ---- normalize and emit ctx (token-major bf16, ready for out-proj) ----
#pragma unroll
  for (int r = 0; r < 8; ++r) {
    float inv = 1.0f / lrow[r];
    int tok = b * SEQ + rowBase + r;
    size_t base = (size_t)tok * DMODEL + (size_t)h * HD + (lane & 15u);
    ctx[base +  0] = f2bf(acc[0][r] * inv);
    ctx[base + 16] = f2bf(acc[1][r] * inv);
    ctx[base + 32] = f2bf(acc[2][r] * inv);
    ctx[base + 48] = f2bf(acc[3][r] * inv);
  }
}

// ---------- launcher ----------

extern "C" void kernel_launch(void* const* d_in, const int* in_sizes, int n_in,
                              void* d_out, int out_size, void* d_ws, size_t ws_size,
                              hipStream_t stream) {
  const float* q    = (const float*)d_in[0];
  const float* k    = (const float*)d_in[1];
  const float* v    = (const float*)d_in[2];
  const int*   mask = (const int*)d_in[3];
  const float* wq   = (const float*)d_in[4];
  const float* wk   = (const float*)d_in[5];
  const float* wv   = (const float*)d_in[6];
  const float* wo   = (const float*)d_in[7];

  char* ws = (char*)d_ws;
  const size_t MB = 1024 * 1024;
  bf16_t* qB  = (bf16_t*)(ws +  0 * MB);   // 8 MB each (4096x1024 bf16)
  bf16_t* kB  = (bf16_t*)(ws +  8 * MB);
  bf16_t* vB  = (bf16_t*)(ws + 16 * MB);
  bf16_t* wqB = (bf16_t*)(ws + 24 * MB);   // 2 MB each (1024x1024 bf16)
  bf16_t* wkB = (bf16_t*)(ws + 26 * MB);
  bf16_t* wvB = (bf16_t*)(ws + 28 * MB);
  bf16_t* woB = (bf16_t*)(ws + 30 * MB);
  bf16_t* Qh  = (bf16_t*)(ws + 32 * MB);   // 8 MB [B,H,S,HD]
  bf16_t* Kh  = (bf16_t*)(ws + 40 * MB);   // 8 MB [B,H,S,HD]
  bf16_t* VT  = (bf16_t*)(ws + 48 * MB);   // 8 MB [B,H,HD,S]
  bf16_t* ctx = (bf16_t*)(ws + 56 * MB);   // 8 MB [NTOK, D]

  const int nAct = NTOK * DMODEL;
  const int nW   = DMODEL * DMODEL;
  cvt_bf16<<<1024, 256, 0, stream>>>(q,  qB,  nAct);
  cvt_bf16<<<1024, 256, 0, stream>>>(k,  kB,  nAct);
  cvt_bf16<<<1024, 256, 0, stream>>>(v,  vB,  nAct);
  cvt_bf16<<<512,  256, 0, stream>>>(wq, wqB, nW);
  cvt_bf16<<<512,  256, 0, stream>>>(wk, wkB, nW);
  cvt_bf16<<<512,  256, 0, stream>>>(wv, wvB, nW);
  cvt_bf16<<<512,  256, 0, stream>>>(wo, woB, nW);

  dim3 gg(NTOK / 128, DMODEL / 64);
  gemm_wmma<<<gg, 128, 0, stream>>>(qB, wqB, Qh, 0);
  gemm_wmma<<<gg, 128, 0, stream>>>(kB, wkB, Kh, 0);
  gemm_wmma<<<gg, 128, 0, stream>>>(vB, wvB, VT, 1);

  attn_wmma<<<dim3(SEQ / 128, NH, NB), 256, 0, stream>>>(Qh, Kh, VT, mask, ctx);

  gemm_wmma<<<gg, 128, 0, stream>>>(ctx, woB, d_out, 2);
}
